// SoftmaxPeepLSTM_63891933495939
// MI455X (gfx1250) — compile-verified, matrix operand order checked by reading, not run on hardware
//
#include <hip/hip_runtime.h>
#include <hip/hip_bf16.h>

// ---------------------------------------------------------------------------
// SoftmaxPeepLSTM on MI455X (gfx1250): bf16 WMMA GEMMs + fused cell update.
// B=64, T=512, D_IN=512, H=1024, 4H=4096, D_OUT=512.
// ---------------------------------------------------------------------------

#define Bc   64
#define Tc   512
#define DINc 512
#define Hc   1024
#define Gc   4096   // 4*H
#define Dc   512    // D_OUT

typedef __bf16 bf16_t;
typedef __attribute__((ext_vector_type(16))) __bf16 v16bf;
typedef __attribute__((ext_vector_type(8)))  float  v8f;

union Frag { v16bf v; uint4 u[2]; };

// Load one 16x32 bf16 WMMA operand fragment for this lane.
// base must point at (row_for_this_lane * ld + k0 + (lane>>4)*8) in a bf16
// row-major matrix. elems 0..7 = K[kb..kb+7], elems 8..15 = K[kb+16..kb+23].
__device__ __forceinline__ v16bf load_frag_g(const bf16_t* base) {
  Frag f;
  const uint4* p = (const uint4*)base;
  f.u[0] = p[0];
  f.u[1] = p[2];   // +16 bf16 elements = +32 bytes
  return f.v;
}

__device__ __forceinline__ v8f wmma_bf16(v16bf a, v16bf b, v8f c) {
  return __builtin_amdgcn_wmma_f32_16x16x32_bf16(false, a, false, b,
                                                 (short)0, c, false, false);
}

__device__ __forceinline__ float sigmoidf_(float x) {
  return 1.0f / (1.0f + __expf(-x));
}

// ---------------------------------------------------------------------------
// Elementwise setup kernels
// ---------------------------------------------------------------------------
__global__ void k_f32_to_bf16(const float* __restrict__ s, bf16_t* __restrict__ d, int n) {
  int i = blockIdx.x * 256 + threadIdx.x;
  if (i < n) d[i] = (bf16_t)s[i];
}

__global__ void k_bias3(const float* __restrict__ bi, const float* __restrict__ bh,
                        const float* __restrict__ bp, float* __restrict__ bhp, int n) {
  int i = blockIdx.x * 256 + threadIdx.x;
  if (i < n) bhp[i] = bi[i] + bh[i] + bp[i];
}

__global__ void k_copy_f32(const float* __restrict__ s, float* __restrict__ d, int n) {
  int i = blockIdx.x * 256 + threadIdx.x;
  if (i < n) d[i] = s[i];
}

// ---------------------------------------------------------------------------
// Phase 1: xg[t][b][g] = sum_k x[b][t][k] * Wi[g][k]   (bias folded into bhp)
// Block: 64 rows (bt) x 64 cols (g); 8 waves: wave w -> mhalf=w&1, ntile=w>>1.
// ---------------------------------------------------------------------------
__global__ void __launch_bounds__(256)
k_gemm_xg(const bf16_t* __restrict__ xb,   // [B*T, DIN] bf16 (row = b*T+t)
          const bf16_t* __restrict__ Wib,  // [G, DIN]  bf16
          float* __restrict__ xg)          // [T, B, G]  fp32
{
  const int wave = threadIdx.x >> 5, lane = threadIdx.x & 31;
  const int ln = lane & 15, hi = lane >> 4;
  const int rowBase = blockIdx.x * 64 + (wave & 1) * 32;
  const int colBase = blockIdx.y * 64 + (wave >> 1) * 16;
  const int g = colBase + ln;

  v8f acc0 = {}, acc1 = {};
  for (int k0 = 0; k0 < DINc; k0 += 32) {
    const int kb = k0 + hi * 8;
    v16bf a0 = load_frag_g(xb + (size_t)(rowBase + ln) * DINc + kb);
    v16bf a1 = load_frag_g(xb + (size_t)(rowBase + 16 + ln) * DINc + kb);
    v16bf b  = load_frag_g(Wib + (size_t)g * DINc + kb);
    acc0 = wmma_bf16(a0, b, acc0);
    acc1 = wmma_bf16(a1, b, acc1);
  }
  for (int r = 0; r < 8; ++r) {
    int row0 = rowBase + r + hi * 8;
    int b0 = row0 >> 9, t0 = row0 & 511;            // row = b*T + t, T=512
    xg[((size_t)t0 * Bc + b0) * Gc + g] = acc0[r];
    int row1 = row0 + 16;
    int b1 = row1 >> 9, t1 = row1 & 511;
    xg[((size_t)t1 * Bc + b1) * Gc + g] = acc1[r];
  }
}

// ---------------------------------------------------------------------------
// Phase 2a: one LSTM step. Each block owns 16 h-indices (all 4 gates) x all
// 64 batch rows, so the cell update fuses in-block via LDS gate exchange.
// 8 waves: gate = w&3, batch-half = w>>2. K = H (Wh x h) then D (Wp x y).
// ---------------------------------------------------------------------------
__global__ void __launch_bounds__(256)
k_lstm_step(const float*  __restrict__ xg,    // [T, B, G]
            const bf16_t* __restrict__ Whb,   // [G, H]
            const bf16_t* __restrict__ Wpb,   // [G, D]
            const float*  __restrict__ bhp,   // [G] = bi+bh+bp
            const bf16_t* __restrict__ hin,   // [B, H]
            const bf16_t* __restrict__ yin,   // [B, D]
            bf16_t*       __restrict__ hout,  // [B, H]
            float*        __restrict__ cst,   // [B, H] (in/out)
            float*        __restrict__ out_h, // [B, T, H]
            int t)
{
  __shared__ float gS[4 * Bc * 16];            // 16 KB: [gate][b][j]

  const int wave = threadIdx.x >> 5, lane = threadIdx.x & 31;
  const int ln = lane & 15, hi = lane >> 4;
  const int gate = wave & 3, mhalf = wave >> 2;
  const int gcol = gate * Hc + blockIdx.x * 16 + ln;  // this lane's G column
  const int r0 = mhalf * 32 + ln;                     // A rows for the 2 m-tiles
  const int r1 = r0 + 16;

  v8f acc0 = {}, acc1 = {};

  // h @ Wh.T over K = H
  for (int k0 = 0; k0 < Hc; k0 += 32) {
    const int kb = k0 + hi * 8;
    v16bf b  = load_frag_g(Whb + (size_t)gcol * Hc + kb);
    v16bf a0 = load_frag_g(hin + (size_t)r0 * Hc + kb);
    v16bf a1 = load_frag_g(hin + (size_t)r1 * Hc + kb);
    acc0 = wmma_bf16(a0, b, acc0);
    acc1 = wmma_bf16(a1, b, acc1);
  }
  // y @ Wp.T over K = D
  for (int k0 = 0; k0 < Dc; k0 += 32) {
    const int kb = k0 + hi * 8;
    v16bf b  = load_frag_g(Wpb + (size_t)gcol * Dc + kb);
    v16bf a0 = load_frag_g(yin + (size_t)r0 * Dc + kb);
    v16bf a1 = load_frag_g(yin + (size_t)r1 * Dc + kb);
    acc0 = wmma_bf16(a0, b, acc0);
    acc1 = wmma_bf16(a1, b, acc1);
  }

  // gates = acc + xg[t] + (bi+bh+bp); exchange through LDS by gate.
  const float bias = bhp[gcol];
  const float* xgt = xg + (size_t)t * Bc * Gc;
  for (int r = 0; r < 8; ++r) {
    int b0 = mhalf * 32 + r + hi * 8;
    gS[gate * (Bc * 16) + b0 * 16 + ln] = acc0[r] + xgt[(size_t)b0 * Gc + gcol] + bias;
    int b1 = b0 + 16;
    gS[gate * (Bc * 16) + b1 * 16 + ln] = acc1[r] + xgt[(size_t)b1 * Gc + gcol] + bias;
  }
  __syncthreads();

  // Fused cell update for this block's 64x16 cells.
  for (int e = threadIdx.x; e < Bc * 16; e += 256) {
    int b = e >> 4, j = e & 15;
    int hh = blockIdx.x * 16 + j;
    float ig = sigmoidf_(gS[0 * (Bc * 16) + b * 16 + j]);
    float fg = sigmoidf_(gS[1 * (Bc * 16) + b * 16 + j]);
    float gg = tanhf   (gS[2 * (Bc * 16) + b * 16 + j]);
    float og = sigmoidf_(gS[3 * (Bc * 16) + b * 16 + j]);
    float c_old = cst[(size_t)b * Hc + hh];
    float cn = fg * c_old + ig * gg;
    cst[(size_t)b * Hc + hh] = cn;
    float hn = og * tanhf(cn);
    out_h[((size_t)b * Tc + t) * Hc + hh] = hn;
    hout[(size_t)b * Hc + hh] = (bf16_t)hn;
  }
}

// ---------------------------------------------------------------------------
// Phase 2b: y = softmax(h @ Wo.T + bo). 4 blocks x 16 batch rows.
// 8 waves, each wave: 1 m-tile x 4 n-tiles (64 cols). Softmax via LDS +
// wave32 shuffle reduction.
// ---------------------------------------------------------------------------
__global__ void __launch_bounds__(256)
k_lstm_softmax(const bf16_t* __restrict__ h,    // [B, H]
               const bf16_t* __restrict__ Wob,  // [D, H]
               const float*  __restrict__ bo,   // [D]
               float*        __restrict__ out_y,// [B, T, D]
               bf16_t*       __restrict__ ybf,  // [B, D]
               int t)
{
  __shared__ float lS[16 * Dc];                 // 32 KB logits

  const int wave = threadIdx.x >> 5, lane = threadIdx.x & 31;
  const int ln = lane & 15, hi = lane >> 4;
  const int rowBase = blockIdx.x * 16;

  v8f acc[4] = {{}, {}, {}, {}};
  for (int k0 = 0; k0 < Hc; k0 += 32) {
    const int kb = k0 + hi * 8;
    v16bf a = load_frag_g(h + (size_t)(rowBase + ln) * Hc + kb);
    for (int nt = 0; nt < 4; ++nt) {
      int g = wave * 64 + nt * 16 + ln;
      v16bf b = load_frag_g(Wob + (size_t)g * Hc + kb);
      acc[nt] = wmma_bf16(a, b, acc[nt]);
    }
  }
  for (int nt = 0; nt < 4; ++nt) {
    int g = wave * 64 + nt * 16 + ln;
    float bias = bo[g];
    for (int r = 0; r < 8; ++r)
      lS[(r + hi * 8) * Dc + g] = acc[nt][r] + bias;
  }
  __syncthreads();

  // 16 rows, 8 waves -> 2 rows per wave; each lane strides 512/32 = 16 elems.
  for (int rr = 0; rr < 2; ++rr) {
    int row = wave * 2 + rr;
    float m = -INFINITY;
    for (int j = lane; j < Dc; j += 32) m = fmaxf(m, lS[row * Dc + j]);
    for (int off = 16; off >= 1; off >>= 1) m = fmaxf(m, __shfl_xor(m, off, 32));
    float s = 0.0f;
    for (int j = lane; j < Dc; j += 32) s += __expf(lS[row * Dc + j] - m);
    for (int off = 16; off >= 1; off >>= 1) s += __shfl_xor(s, off, 32);
    float inv = 1.0f / s;
    int b = rowBase + row;
    for (int j = lane; j < Dc; j += 32) {
      float yv = __expf(lS[row * Dc + j] - m) * inv;
      out_y[((size_t)b * Tc + t) * Dc + j] = yv;
      ybf[(size_t)b * Dc + j] = (bf16_t)yv;
    }
  }
}

// ---------------------------------------------------------------------------
// Final states: hT = out0[:,T-1,:], cT = c, yT = out1[:,T-1,:]
// ---------------------------------------------------------------------------
__global__ void k_finalize(const float* __restrict__ out0, const float* __restrict__ out1,
                           const float* __restrict__ c,
                           float* __restrict__ hT, float* __restrict__ cT,
                           float* __restrict__ yT)
{
  int i = blockIdx.x * 256 + threadIdx.x;
  if (i < Bc * Hc) {
    int b = i >> 10, hh = i & 1023;
    hT[i] = out0[((size_t)b * Tc + (Tc - 1)) * Hc + hh];
    cT[i] = c[i];
  }
  if (i < Bc * Dc) {
    int b = i >> 9, n = i & 511;
    yT[i] = out1[((size_t)b * Tc + (Tc - 1)) * Dc + n];
  }
}

// ---------------------------------------------------------------------------
extern "C" void kernel_launch(void* const* d_in, const int* in_sizes, int n_in,
                              void* d_out, int out_size, void* d_ws, size_t ws_size,
                              hipStream_t stream) {
  const float* x  = (const float*)d_in[0];
  const float* hx = (const float*)d_in[1];
  const float* cx = (const float*)d_in[2];
  const float* yx = (const float*)d_in[3];
  const float* Wi = (const float*)d_in[4];
  const float* bi = (const float*)d_in[5];
  const float* Wh = (const float*)d_in[6];
  const float* bh = (const float*)d_in[7];
  const float* Wp = (const float*)d_in[8];
  const float* bp = (const float*)d_in[9];
  const float* Wo = (const float*)d_in[10];
  const float* bo = (const float*)d_in[11];

  float* out = (float*)d_out;
  float* out0 = out;                                   // [B,T,H]
  float* out1 = out + (size_t)Bc * Tc * Hc;            // [B,T,D]
  float* out2 = out1 + (size_t)Bc * Tc * Dc;           // hT [B,H]
  float* out3 = out2 + (size_t)Bc * Hc;                // cT [B,H]
  float* out4 = out3 + (size_t)Bc * Hc;                // yT [B,D]

  // Workspace carve-up
  char* ws = (char*)d_ws;
  float*  xg   = (float*)ws;   ws += (size_t)Tc * Bc * Gc * 4;   // 512 MB
  bf16_t* xbf  = (bf16_t*)ws;  ws += (size_t)Bc * Tc * DINc * 2; // 32 MB
  bf16_t* Whb  = (bf16_t*)ws;  ws += (size_t)Gc * Hc * 2;        // 8 MB
  bf16_t* Wpb  = (bf16_t*)ws;  ws += (size_t)Gc * Dc * 2;        // 4 MB
  bf16_t* Wib  = (bf16_t*)ws;  ws += (size_t)Gc * DINc * 2;      // 4 MB
  bf16_t* Wob  = (bf16_t*)ws;  ws += (size_t)Dc * Hc * 2;        // 1 MB
  float*  bhp  = (float*)ws;   ws += (size_t)Gc * 4;             // 16 KB
  bf16_t* hbf0 = (bf16_t*)ws;  ws += (size_t)Bc * Hc * 2;        // 128 KB
  bf16_t* hbf1 = (bf16_t*)ws;  ws += (size_t)Bc * Hc * 2;        // 128 KB
  bf16_t* ybf  = (bf16_t*)ws;  ws += (size_t)Bc * Dc * 2;        // 64 KB
  float*  cst  = (float*)ws;   ws += (size_t)Bc * Hc * 4;        // 256 KB
  bf16_t* hbufs[2] = {hbf0, hbf1};

  // --- setup: conversions, fused bias, c-state init ---
  auto cv = [&](const float* s, bf16_t* d, int n) {
    k_f32_to_bf16<<<(n + 255) / 256, 256, 0, stream>>>(s, d, n);
  };
  cv(x,  xbf, Bc * Tc * DINc);
  cv(Wi, Wib, Gc * DINc);
  cv(Wh, Whb, Gc * Hc);
  cv(Wp, Wpb, Gc * Dc);
  cv(Wo, Wob, Dc * Hc);
  cv(hx, hbf0, Bc * Hc);
  cv(yx, ybf,  Bc * Dc);
  k_bias3<<<Gc / 256, 256, 0, stream>>>(bi, bh, bp, bhp, Gc);
  k_copy_f32<<<(Bc * Hc + 255) / 256, 256, 0, stream>>>(cx, cst, Bc * Hc);

  // --- phase 1: input projection GEMM ---
  dim3 g0((Bc * Tc) / 64, Gc / 64);
  k_gemm_xg<<<g0, 256, 0, stream>>>(xbf, Wib, xg);

  // --- phase 2: sequential scan, 2 kernels per step ---
  for (int t = 0; t < Tc; ++t) {
    const bf16_t* hin = hbufs[t & 1];
    bf16_t* hnext = hbufs[(t + 1) & 1];
    k_lstm_step<<<Hc / 16, 256, 0, stream>>>(xg, Whb, Wpb, bhp, hin, ybf,
                                             hnext, cst, out0, t);
    k_lstm_softmax<<<Bc / 16, 256, 0, stream>>>(hnext, Wob, bo, out1, ybf, t);
  }

  // --- final states ---
  k_finalize<<<(Bc * Hc + 255) / 256, 256, 0, stream>>>(out0, out1, cst,
                                                        out2, out3, out4);
}
